// MultiHeadAttention_29008209117372
// MI455X (gfx1250) — matrix operand order repaired and checked by reading, not compiled, not run on hardware
//
#include <hip/hip_runtime.h>
#include <hip/hip_bf16.h>

// ---------------------------------------------------------------------------
// MHA forward for MI455X (gfx1250, wave32): bf16 WMMA compute, fp32 accum.
//   1) convert fp32 inputs/weights -> bf16
//   2) q/k projections -> [b,h,s,64] bf16 ; v projection -> transposed [b,h,64,s]
//   3) flash-style attention (one 16-row query tile per wave, online softmax)
//   4) output projection -> fp32 d_out
// ---------------------------------------------------------------------------

#define DMODEL 1024
#define NHEADS 16
#define DK 64
#define BATCH 4
#define SEQ 2048

typedef __attribute__((ext_vector_type(16))) __bf16 v16bf;
typedef __attribute__((ext_vector_type(8)))  float  v8f;

union ABReg {
  uint4 q[2];   // 8 VGPRs = 16 packed bf16
  v16bf bf;
};

__device__ __forceinline__ unsigned short f2bf(float f) {
  unsigned int u = __builtin_bit_cast(unsigned int, f);
  u += 0x7FFFu + ((u >> 16) & 1u);           // round-to-nearest-even
  return (unsigned short)(u >> 16);
}

// ------------------------- fp32 -> bf16 conversion -------------------------
__global__ __launch_bounds__(256) void cvt_f32_to_bf16(
    const float* __restrict__ in, unsigned int* __restrict__ out, int n2) {
  int i = blockIdx.x * blockDim.x + threadIdx.x;
  if (i < n2) {
    float a = in[2 * i], b = in[2 * i + 1];
    out[i] = (unsigned int)f2bf(a) | ((unsigned int)f2bf(b) << 16);
  }
}

// ------------------------------- GEMM ---------------------------------------
// out = X(MxK) @ W(NxK)^T + bias.  M=8192, N=K=1024.
// mode 0/1: bf16 out in q/k head layout [b,h,s,64]
// mode 2  : bf16 out transposed        [b,h,64,s]
// mode 3  : fp32 out row-major         [m, n]   (final projection)
// Block: 256 thr (8 waves). Block tile 128x128; wave tile 64x32 (4x2 WMMA accs).
__global__ __launch_bounds__(256) void gemm_bf16(
    const unsigned short* __restrict__ X, const unsigned short* __restrict__ W,
    const float* __restrict__ bias, unsigned short* __restrict__ out_bf,
    float* __restrict__ out_f, int mode) {
  const int lane = threadIdx.x & 31;
  const int wave = threadIdx.x >> 5;
  const int hgrp = lane >> 4;     // half-wave select
  const int lo   = lane & 15;
  const int rowBase = blockIdx.y * 128 + (wave >> 2) * 64;
  const int colBase = blockIdx.x * 128 + (wave & 3) * 32;

  v8f acc[4][2];
#pragma unroll
  for (int i = 0; i < 4; ++i)
#pragma unroll
    for (int j = 0; j < 2; ++j)
#pragma unroll
      for (int r = 0; r < 8; ++r) acc[i][j][r] = 0.0f;

  for (int kk = 0; kk < DMODEL; kk += 32) {
    ABReg A[4];
#pragma unroll
    for (int rt = 0; rt < 4; ++rt) {
      const unsigned short* ap =
          X + (size_t)(rowBase + rt * 16 + lo) * DMODEL + kk + hgrp * 8;
      A[rt].q[0] = *reinterpret_cast<const uint4*>(ap);
      A[rt].q[1] = *reinterpret_cast<const uint4*>(ap + 16);
    }
    ABReg B[2];
#pragma unroll
    for (int ct = 0; ct < 2; ++ct) {
      const unsigned short* bp =
          W + (size_t)(colBase + ct * 16 + lo) * DMODEL + kk + hgrp * 8;
      B[ct].q[0] = *reinterpret_cast<const uint4*>(bp);
      B[ct].q[1] = *reinterpret_cast<const uint4*>(bp + 16);
    }
#pragma unroll
    for (int rt = 0; rt < 4; ++rt)
#pragma unroll
      for (int ct = 0; ct < 2; ++ct)
        acc[rt][ct] = __builtin_amdgcn_wmma_f32_16x16x32_bf16(
            false, A[rt].bf, false, B[ct].bf, (short)0, acc[rt][ct], false,
            false);
  }

  // epilogue: C layout = (lane lo -> col, VGPR r -> row r + 8*hgrp)
#pragma unroll
  for (int rt = 0; rt < 4; ++rt) {
#pragma unroll
    for (int ct = 0; ct < 2; ++ct) {
      const int n = colBase + ct * 16 + lo;
      const float bv = bias[n];
#pragma unroll
      for (int r = 0; r < 8; ++r) {
        const int m = rowBase + rt * 16 + r + 8 * hgrp;
        const float v = acc[rt][ct][r] + bv;
        if (mode == 3) {
          out_f[(size_t)m * DMODEL + n] = v;
        } else {
          const int b = m >> 11, s = m & (SEQ - 1);
          const int h = n >> 6, d = n & (DK - 1);
          if (mode == 2)  // v transposed: [b,h,d,s]
            out_bf[(((size_t)(b * NHEADS + h) * DK + d) << 11) + s] = f2bf(v);
          else            // q/k: [b,h,s,d]
            out_bf[((size_t)(b * NHEADS + h) * SEQ + s) * DK + d] = f2bf(v);
        }
      }
    }
  }
}

// ----------------------------- attention ------------------------------------
// One wave per 16-row query tile; streams causal 32-wide key chunks.
__global__ __launch_bounds__(256) void attn_flash(
    const unsigned short* __restrict__ qh, const unsigned short* __restrict__ kh,
    const unsigned short* __restrict__ vt, unsigned short* __restrict__ ctx) {
  __shared__ __align__(16) unsigned short plds[8][512];  // per-wave 16x32 bf16
  const int lane = threadIdx.x & 31;
  const int wave = threadIdx.x >> 5;
  const int hgrp = lane >> 4;
  const int lo   = lane & 15;
  const int tile = blockIdx.x * 8 + wave;  // 0..8191
  const int bh = tile >> 7;                // b*NHEADS + h
  const int qbase = (tile & 127) * 16;
  const int b = bh >> 4, h = bh & 15;

  const unsigned short* qp = qh + (size_t)bh * SEQ * DK;
  const unsigned short* kp = kh + (size_t)bh * SEQ * DK;
  const unsigned short* vp = vt + (size_t)bh * DK * SEQ;

  // preload Q tile (16 x 64) as two 16x32 A fragments
  ABReg AQ[2];
#pragma unroll
  for (int j = 0; j < 2; ++j) {
    const unsigned short* ap = qp + (size_t)(qbase + lo) * DK + j * 32 + hgrp * 8;
    AQ[j].q[0] = *reinterpret_cast<const uint4*>(ap);
    AQ[j].q[1] = *reinterpret_cast<const uint4*>(ap + 16);
  }

  v8f acc[4];
#pragma unroll
  for (int t = 0; t < 4; ++t)
#pragma unroll
    for (int r = 0; r < 8; ++r) acc[t][r] = 0.0f;
  float mrow[8], lrow[8];
#pragma unroll
  for (int r = 0; r < 8; ++r) { mrow[r] = -__builtin_inff(); lrow[r] = 0.0f; }

  const float scale = 0.125f;  // 1/sqrt(64)
  const int qmax = qbase + 15;

  for (int jbase = 0; jbase <= qmax; jbase += 32) {
    v8f st[2];
#pragma unroll
    for (int ct = 0; ct < 2; ++ct) {
      const int j0 = jbase + ct * 16 + lo;  // this lane's key column
      ABReg BK0, BK1;
      const unsigned short* bp = kp + (size_t)j0 * DK + hgrp * 8;
      BK0.q[0] = *reinterpret_cast<const uint4*>(bp);
      BK0.q[1] = *reinterpret_cast<const uint4*>(bp + 16);
      BK1.q[0] = *reinterpret_cast<const uint4*>(bp + 32);
      BK1.q[1] = *reinterpret_cast<const uint4*>(bp + 48);
      v8f s;
#pragma unroll
      for (int r = 0; r < 8; ++r) s[r] = 0.0f;
      s = __builtin_amdgcn_wmma_f32_16x16x32_bf16(false, AQ[0].bf, false,
                                                  BK0.bf, (short)0, s, false,
                                                  false);
      s = __builtin_amdgcn_wmma_f32_16x16x32_bf16(false, AQ[1].bf, false,
                                                  BK1.bf, (short)0, s, false,
                                                  false);
#pragma unroll
      for (int r = 0; r < 8; ++r) {
        const int row = qbase + r + 8 * hgrp;
        float v = s[r] * scale;
        if (j0 > row) v = -__builtin_inff();  // causal mask
        s[r] = v;
      }
      st[ct] = s;
    }

    // online softmax; rows live on 16-lane half-groups -> xor reduce 1,2,4,8
#pragma unroll
    for (int r = 0; r < 8; ++r) {
      float cm = fmaxf(st[0][r], st[1][r]);
      cm = fmaxf(cm, __shfl_xor(cm, 1));
      cm = fmaxf(cm, __shfl_xor(cm, 2));
      cm = fmaxf(cm, __shfl_xor(cm, 4));
      cm = fmaxf(cm, __shfl_xor(cm, 8));
      const float mnew = fmaxf(mrow[r], cm);
      const float p0 = __expf(st[0][r] - mnew);
      const float p1 = __expf(st[1][r] - mnew);
      float rs = p0 + p1;
      rs += __shfl_xor(rs, 1);
      rs += __shfl_xor(rs, 2);
      rs += __shfl_xor(rs, 4);
      rs += __shfl_xor(rs, 8);
      const float sf = __expf(mrow[r] - mnew);
      lrow[r] = lrow[r] * sf + rs;
      mrow[r] = mnew;
#pragma unroll
      for (int t = 0; t < 4; ++t) acc[t][r] = acc[t][r] * sf;
      const int row_l = r + 8 * hgrp;
      plds[wave][row_l * 32 + lo]      = f2bf(p0);  // C-layout -> LDS row-major
      plds[wave][row_l * 32 + 16 + lo] = f2bf(p1);
    }

    // reload P as A-matrix fragment (same wave; LDS ops are in-order)
    ABReg AP;
    {
      const unsigned short* pp = &plds[wave][lo * 32 + hgrp * 8];
      AP.q[0] = *reinterpret_cast<const uint4*>(pp);
      AP.q[1] = *reinterpret_cast<const uint4*>(pp + 16);
    }
    // O += P @ V_chunk ; vt is [d, s] so B fragments are contiguous in s
#pragma unroll
    for (int td = 0; td < 4; ++td) {
      const int d = td * 16 + lo;
      ABReg BV;
      const unsigned short* bp = vp + (size_t)d * SEQ + jbase + hgrp * 8;
      BV.q[0] = *reinterpret_cast<const uint4*>(bp);
      BV.q[1] = *reinterpret_cast<const uint4*>(bp + 16);
      acc[td] = __builtin_amdgcn_wmma_f32_16x16x32_bf16(
          false, AP.bf, false, BV.bf, (short)0, acc[td], false, false);
    }
  }

  // normalize and write ctx [b, s, h*64 + d] bf16 for final projection
#pragma unroll
  for (int r = 0; r < 8; ++r) {
    const float inv = 1.0f / lrow[r];
    const int row = qbase + r + 8 * hgrp;
#pragma unroll
    for (int td = 0; td < 4; ++td) {
      ctx[(size_t)(b * SEQ + row) * DMODEL + h * DK + td * 16 + lo] =
          f2bf(acc[td][r] * inv);
    }
  }
}

// ------------------------------ launcher ------------------------------------
extern "C" void kernel_launch(void* const* d_in, const int* in_sizes, int n_in,
                              void* d_out, int out_size, void* d_ws,
                              size_t ws_size, hipStream_t stream) {
  (void)in_sizes; (void)n_in; (void)out_size; (void)ws_size;
  const float* Q  = (const float*)d_in[0];
  const float* K  = (const float*)d_in[1];
  const float* V  = (const float*)d_in[2];
  /* d_in[3] = mask: causality is hardcoded in attn_flash */
  const float* Wq = (const float*)d_in[4];
  const float* bq = (const float*)d_in[5];
  const float* Wk = (const float*)d_in[6];
  const float* bk = (const float*)d_in[7];
  const float* Wv = (const float*)d_in[8];
  const float* bv = (const float*)d_in[9];
  const float* Wo = (const float*)d_in[10];
  const float* bo = (const float*)d_in[11];

  const size_t ACT = (size_t)BATCH * SEQ * DMODEL;  // 8.39M elems
  const size_t WEI = (size_t)DMODEL * DMODEL;       // 1.05M elems

  char* ws = (char*)d_ws;
  unsigned short* Qb  = (unsigned short*)ws; ws += ACT * 2;
  unsigned short* Kb  = (unsigned short*)ws; ws += ACT * 2;
  unsigned short* Vb  = (unsigned short*)ws; ws += ACT * 2;
  unsigned short* Wqb = (unsigned short*)ws; ws += WEI * 2;
  unsigned short* Wkb = (unsigned short*)ws; ws += WEI * 2;
  unsigned short* Wvb = (unsigned short*)ws; ws += WEI * 2;
  unsigned short* Wob = (unsigned short*)ws; ws += WEI * 2;
  unsigned short* qh  = (unsigned short*)ws; ws += ACT * 2;
  unsigned short* kh  = (unsigned short*)ws; ws += ACT * 2;
  unsigned short* vtw = (unsigned short*)ws; ws += ACT * 2;
  unsigned short* ctx = (unsigned short*)ws; ws += ACT * 2;

  // 1) conversions
  {
    int n2a = (int)(ACT / 2), n2w = (int)(WEI / 2);
    int ga = (n2a + 255) / 256, gw = (n2w + 255) / 256;
    cvt_f32_to_bf16<<<ga, 256, 0, stream>>>(Q,  (unsigned int*)Qb,  n2a);
    cvt_f32_to_bf16<<<ga, 256, 0, stream>>>(K,  (unsigned int*)Kb,  n2a);
    cvt_f32_to_bf16<<<ga, 256, 0, stream>>>(V,  (unsigned int*)Vb,  n2a);
    cvt_f32_to_bf16<<<gw, 256, 0, stream>>>(Wq, (unsigned int*)Wqb, n2w);
    cvt_f32_to_bf16<<<gw, 256, 0, stream>>>(Wk, (unsigned int*)Wkb, n2w);
    cvt_f32_to_bf16<<<gw, 256, 0, stream>>>(Wv, (unsigned int*)Wvb, n2w);
    cvt_f32_to_bf16<<<gw, 256, 0, stream>>>(Wo, (unsigned int*)Wob, n2w);
  }

  // 2) projections (M=8192, N=1024 -> grid 8 x 64, 256 thr)
  dim3 ggrid(DMODEL / 128, (BATCH * SEQ) / 128);
  gemm_bf16<<<ggrid, 256, 0, stream>>>(Qb, Wqb, bq, qh,  nullptr, 0);
  gemm_bf16<<<ggrid, 256, 0, stream>>>(Kb, Wkb, bk, kh,  nullptr, 1);
  gemm_bf16<<<ggrid, 256, 0, stream>>>(Vb, Wvb, bv, vtw, nullptr, 2);

  // 3) attention: 64 (b,h) x 128 q-tiles = 8192 wave tiles / 8 waves
  attn_flash<<<1024, 256, 0, stream>>>(qh, kh, vtw, ctx);

  // 4) output projection -> fp32 d_out
  gemm_bf16<<<ggrid, 256, 0, stream>>>(ctx, Wob, bo, nullptr, (float*)d_out, 3);
}
